// VectorQuantizer_84250078478369
// MI455X (gfx1250) — matrix operand order repaired and checked by reading, not compiled, not run on hardware
//
#include <hip/hip_runtime.h>
#include <hip/hip_bf16.h>

typedef __attribute__((ext_vector_type(16))) _Float16 v16h;
typedef __attribute__((ext_vector_type(8)))  float    v8f;
typedef __attribute__((ext_vector_type(4)))  int      v4i;

#define AS1 __attribute__((address_space(1)))
#define AS3 __attribute__((address_space(3)))

#if __has_builtin(__builtin_amdgcn_global_load_async_to_lds_b128)
#define HAVE_ASYNC_LDS 1
#endif

// Problem constants
#define N_E   1024
#define E_DIM 256
#define HW    4096            // 64*64
#define NVEC  65536           // 16*64*64 rows
#define NT    64              // N-tiles of 16 codes
#define KSTEPS 8              // 256 / 32
#define OUT_ELEMS 16777216    // 16*256*64*64

// B staging: chunks of 4 N-tiles, double buffered in LDS
#define CHUNK_NT    4
#define NCHUNK      (NT / CHUNK_NT)                  // 16
#define CHUNK_BYTES (CHUNK_NT * KSTEPS * 32 * 32)    // 32768
#define COPIES      (CHUNK_BYTES / 16 / 256)         // 8 x b128 per thread

// Workspace layout (bytes)
#define WS_BPACK   0          // NT*KSTEPS*32 lanes * 16 halves = 524288 B
#define WS_NORMS   524288     // 1024 f32
#define WS_IDX     528384     // 65536 i32
#define WS_PART    790528     // 65536 f32 partial sums

// ---------------------------------------------------------------------------
// Kernel 1: pack codebook (fp32 -> f16) into the WMMA B-matrix register image.
// B layout (16-bit, 32x16, contiguous k as packed here): lane L holds column N = L%16;
// lanes 0-15 hold K=0..15 (VGPR v = K 2v,2v+1), lanes 16-31 hold K=16..31.
// ---------------------------------------------------------------------------
__global__ __launch_bounds__(256) void vq_pack_b(const float* __restrict__ cb,
                                                 _Float16* __restrict__ bp) {
  int tid = blockIdx.x * 256 + threadIdx.x;   // 0 .. 262143
  int e    = tid & 15;
  int lane = (tid >> 4) & 31;
  int ks   = (tid >> 9) & 7;
  int nt   = tid >> 12;
  int n    = nt * 16 + (lane & 15);
  int k    = ((lane >> 4) << 4) + e;          // 16*half + e
  bp[tid]  = (_Float16)cb[n * E_DIM + ks * 32 + k];
}

// ---------------------------------------------------------------------------
// Kernel 2: per-code squared norms ||e_n||^2
// ---------------------------------------------------------------------------
__global__ __launch_bounds__(256) void vq_norms(const float* __restrict__ cb,
                                                float* __restrict__ norms) {
  int n = blockIdx.x * 256 + threadIdx.x;
  if (n < N_E) {
    float s = 0.f;
    const float* row = cb + n * E_DIM;
#pragma unroll 8
    for (int k = 0; k < E_DIM; ++k) s += row[k] * row[k];
    norms[n] = s;
  }
}

// ---------------------------------------------------------------------------
// Kernel 3: fused GEMM (z . cb^T) + argmin.
// Block = 8 waves; each wave owns TWO 16-row M tiles (32 rows) -> 256 rows
// per block, 256 blocks. B tiles are staged into LDS by the whole block
// (async global->LDS, double buffered) and consumed by all 8 waves, so
// each B element crosses L2->WGP once per block instead of once per wave.
// ---------------------------------------------------------------------------
__global__ __launch_bounds__(256) void vq_gemm_argmin(const float* __restrict__ z,
                                                      const v4i* __restrict__ bp4,
                                                      const float* __restrict__ norms,
                                                      int* __restrict__ idxout) {
  extern __shared__ __attribute__((aligned(32))) char smem[];  // 2*CHUNK_BYTES
  const int wave  = threadIdx.x >> 5;
  const int lane  = threadIdx.x & 31;
  const int laneN = lane & 15;
  const int half  = lane >> 4;
  const int rowBase = (blockIdx.x * 8 + wave) * 32;

  // Load two 16x256 A tiles (fp32 -> f16) into registers, contiguous-k A layout (same k order as B):
  // lane L holds row M=L%16; lanes 0-15: K0..15, lanes 16-31: K16..31.
  v16h A0[KSTEPS], A1[KSTEPS];
  {
    const int row0 = rowBase + laneN;
    const int row1 = row0 + 16;
    const float* zr0 = z + (size_t)(row0 >> 12) * (E_DIM * HW) + (row0 & 4095);
    const float* zr1 = z + (size_t)(row1 >> 12) * (E_DIM * HW) + (row1 & 4095);
#pragma unroll
    for (int ks = 0; ks < KSTEPS; ++ks) {
#pragma unroll
      for (int v = 0; v < 8; ++v) {
        int kk = 2 * v + 16 * half;
        size_t c0 = (size_t)(ks * 32 + kk) * HW;
        A0[ks][2 * v]     = (_Float16)zr0[c0];
        A0[ks][2 * v + 1] = (_Float16)zr0[c0 + HW];
        A1[ks][2 * v]     = (_Float16)zr1[c0];
        A1[ks][2 * v + 1] = (_Float16)zr1[c0 + HW];
      }
    }
  }

  // Cooperative B-chunk stage: 256 threads x 8 async b128 = 32 KB per chunk.
  auto issue = [&](int c, int buf) {
    const v4i* g = bp4 + (size_t)c * (CHUNK_BYTES / 16);
#ifdef HAVE_ASYNC_LDS
    AS3 v4i* l = (AS3 v4i*)(smem + buf * CHUNK_BYTES);
#pragma unroll
    for (int i = 0; i < COPIES; ++i) {
      int e2 = threadIdx.x + i * 256;
      __builtin_amdgcn_global_load_async_to_lds_b128(
          (AS1 v4i*)(g + e2), (AS3 v4i*)(l + e2), 0, 0);
    }
#else
    v4i* l = (v4i*)(smem + buf * CHUNK_BYTES);
#pragma unroll
    for (int i = 0; i < COPIES; ++i) {
      int e2 = threadIdx.x + i * 256;
      l[e2] = g[e2];
    }
#endif
  };
  auto waitAsync = [] {
#ifdef HAVE_ASYNC_LDS
#if __has_builtin(__builtin_amdgcn_s_wait_asynccnt)
    __builtin_amdgcn_s_wait_asynccnt(0);
#else
    asm volatile("s_wait_asynccnt 0x0" ::: "memory");
#endif
#endif
  };

  float best0[8], best1[8];
  int   bn0[8], bn1[8];
#pragma unroll
  for (int v = 0; v < 8; ++v) {
    best0[v] = 3.4e38f; best1[v] = 3.4e38f; bn0[v] = 0; bn1[v] = 0;
  }

  issue(0, 0);
  for (int c = 0; c < NCHUNK; ++c) {
    const int cur = c & 1;
    waitAsync();              // chunk c resident (in-order async completion)
    __syncthreads();          // visible to all waves
    if (c + 1 < NCHUNK) issue(c + 1, cur ^ 1);  // overlap DMA with WMMA

    const v16h* bt = (const v16h*)(smem + cur * CHUNK_BYTES);
#pragma unroll
    for (int ntl = 0; ntl < CHUNK_NT; ++ntl) {
      const int n = (c * CHUNK_NT + ntl) * 16 + laneN;
      const float nrm = norms[n];             // ||e_N||^2 for this lane's N
      v8f acc0 = {}, acc1 = {};
#pragma unroll
      for (int ks = 0; ks < KSTEPS; ++ks) {
        v16h B = bt[(ntl * KSTEPS + ks) * 32 + lane];   // ds_load_b128 x2
        acc0 = __builtin_amdgcn_wmma_f32_16x16x32_f16(
            false, A0[ks], false, B, (short)0, acc0, false, false);
        acc1 = __builtin_amdgcn_wmma_f32_16x16x32_f16(
            false, A1[ks], false, B, (short)0, acc1, false, false);
      }
#pragma unroll
      for (int v = 0; v < 8; ++v) {
        float s0 = nrm - 2.0f * acc0[v];      // ||e||^2 - 2 z.e
        float s1 = nrm - 2.0f * acc1[v];
        if (s0 < best0[v]) { best0[v] = s0; bn0[v] = n; }
        if (s1 < best1[v]) { best1[v] = s1; bn1[v] = n; }
      }
    }
    __syncthreads();          // buffer consumed before it is refilled
  }

  // Reduce over N: 16 lanes per half-wave hold the 16 columns of each row.
#pragma unroll
  for (int v = 0; v < 8; ++v) {
    float v0 = best0[v], v1 = best1[v];
    int   n0 = bn0[v],   n1 = bn1[v];
#pragma unroll
    for (int off = 8; off >= 1; off >>= 1) {
      float o0 = __shfl_xor(v0, off, 32); int m0 = __shfl_xor(n0, off, 32);
      float o1 = __shfl_xor(v1, off, 32); int m1 = __shfl_xor(n1, off, 32);
      if (o0 < v0 || (o0 == v0 && m0 < n0)) { v0 = o0; n0 = m0; }
      if (o1 < v1 || (o1 == v1 && m1 < n1)) { v1 = o1; n1 = m1; }
    }
    if (laneN == 0) {
      idxout[rowBase + v + 8 * half]      = n0;
      idxout[rowBase + 16 + v + 8 * half] = n1;
    }
  }
}

// ---------------------------------------------------------------------------
// Kernel 4: gather z_q (NCHW), per-block partial sums of (z_q - z)^2.
// ---------------------------------------------------------------------------
__global__ __launch_bounds__(256) void vq_gather(const float* __restrict__ z,
                                                 const float* __restrict__ cb,
                                                 const int* __restrict__ idx,
                                                 float* __restrict__ out,
                                                 float* __restrict__ partial) {
  int i = blockIdx.x * 256 + threadIdx.x;     // flat NCHW index
  int c    = (i >> 12) & 255;
  int bimg = i >> 20;
  int hw   = i & 4095;
  int code = idx[(bimg << 12) | hw];
  float q  = cb[code * E_DIM + c];
  float zv = z[i];
  out[i]   = q;
  float d  = q - zv;

  __shared__ float red[256];
  red[threadIdx.x] = d * d;
  __syncthreads();
#pragma unroll
  for (int s = 128; s > 0; s >>= 1) {
    if (threadIdx.x < s) red[threadIdx.x] += red[threadIdx.x + s];
    __syncthreads();
  }
  if (threadIdx.x == 0) partial[blockIdx.x] = red[0];
}

// ---------------------------------------------------------------------------
// Kernel 5: deterministic final reduction -> vq_loss = 1.25 * mean
// ---------------------------------------------------------------------------
__global__ __launch_bounds__(256) void vq_final(const float* __restrict__ partial,
                                                float* __restrict__ loss) {
  float s = 0.f;
  for (int i = threadIdx.x; i < NVEC; i += 256) s += partial[i];
  __shared__ float red[256];
  red[threadIdx.x] = s;
  __syncthreads();
#pragma unroll
  for (int st = 128; st > 0; st >>= 1) {
    if (threadIdx.x < st) red[threadIdx.x] += red[threadIdx.x + st];
    __syncthreads();
  }
  if (threadIdx.x == 0)
    loss[0] = red[0] * (1.25f / (float)OUT_ELEMS);  // (1+beta)*mean
}

extern "C" void kernel_launch(void* const* d_in, const int* in_sizes, int n_in,
                              void* d_out, int out_size, void* d_ws, size_t ws_size,
                              hipStream_t stream) {
  const float* z  = (const float*)d_in[0];   // [16,256,64,64]
  const float* cb = (const float*)d_in[1];   // [1024,256]
  float* out  = (float*)d_out;               // [16,256,64,64] then loss scalar
  float* loss = out + OUT_ELEMS;

  char* ws = (char*)d_ws;
  _Float16* bpack = (_Float16*)(ws + WS_BPACK);
  float*    norms = (float*)(ws + WS_NORMS);
  int*      idx   = (int*)(ws + WS_IDX);
  float*    part  = (float*)(ws + WS_PART);

  vq_pack_b<<<262144 / 256, 256, 0, stream>>>(cb, bpack);
  vq_norms<<<N_E / 256, 256, 0, stream>>>(cb, norms);
  vq_gemm_argmin<<<NVEC / 256, 256, 2 * CHUNK_BYTES, stream>>>(
      z, (const v4i*)bpack, norms, idx);
  vq_gather<<<OUT_ELEMS / 256, 256, 0, stream>>>(z, cb, idx, out, part);
  vq_final<<<1, 256, 0, stream>>>(part, loss);
}
